// SelfAttention_20864951124606
// MI455X (gfx1250) — matrix operand order repaired and checked by reading, not compiled, not run on hardware
//
#include <hip/hip_runtime.h>

// ---------------------------------------------------------------------------
// Types for WMMA bf16
// ---------------------------------------------------------------------------
typedef __attribute__((ext_vector_type(16))) __bf16 v16bf;
typedef __attribute__((ext_vector_type(8)))  __bf16 v8bf;
typedef __attribute__((ext_vector_type(4)))  __bf16 v4bf;
typedef __attribute__((ext_vector_type(8)))  float  v8f;

#define DI __device__ __forceinline__

DI v8f wmma_bf16(v16bf a, v16bf b, v8f c) {
  // D = A(16x32) * B(32x16) + C ; f32 accumulate
  return __builtin_amdgcn_wmma_f32_16x16x32_bf16(
      /*neg_a=*/false, a, /*neg_b=*/false, b,
      /*c_mod=*/(short)0, c, /*reuse_a=*/false, /*reuse_b=*/false);
}

// A-fragment (16x32 bf16, row-major source, row = lane&15):
// elements 0..15 -> K = kbase+0..15
// caller passes p = &buf[row*ld + k0 + kbase] with kbase = (lane<16)?0:16
DI v16bf load_a_frag(const __bf16* p) {
  v8bf lo = *(const v8bf*)(p);
  v8bf hi = *(const v8bf*)(p + 8);
  v16bf r;
#pragma unroll
  for (int i = 0; i < 8; ++i) { r[i] = lo[i]; r[i + 8] = hi[i]; }
  return r;
}

// Same A-fragment but source is f32 (softmax probabilities in LDS)
DI v16bf load_a_frag_f32(const float* p) {
  v16bf r;
#pragma unroll
  for (int i = 0; i < 8; ++i) {
    r[i]     = (__bf16)p[i];
    r[i + 8] = (__bf16)p[8 + i];
  }
  return r;
}

// ---------------------------------------------------------------------------
// Async global->LDS staging (CDNA5 GLOBAL_LOAD_ASYNC_TO_LDS_B128, ASYNCcnt)
// Builtin exists on this toolchain; parameter 0 is `int4 addrspace(1)*`
// (per the round-2 diagnostic), so route both pointers through typed,
// address-space-qualified int4 pointers.
// ---------------------------------------------------------------------------
#if defined(__AMDGCN__) && __has_builtin(__builtin_amdgcn_global_load_async_to_lds_b128) && __has_builtin(__builtin_amdgcn_s_wait_asynccnt)
#define USE_ASYNC_LDS 1
#else
#define USE_ASYNC_LDS 0
#endif

#if USE_ASYNC_LDS
typedef __attribute__((ext_vector_type(4))) int v4i;
typedef __attribute__((address_space(1))) v4i gv4i_t;   // global int4
typedef __attribute__((address_space(3))) v4i lv4i_t;   // LDS int4
#endif

DI void async_copy_b128(void* lds_dst, const void* gsrc) {
#if USE_ASYNC_LDS
  __builtin_amdgcn_global_load_async_to_lds_b128(
      (gv4i_t*)(unsigned long long)gsrc,
      (lv4i_t*)(unsigned int)(unsigned long long)lds_dst,
      /*offset=*/0, /*cpol=*/0);
#else
  *(v8bf*)lds_dst = *(const v8bf*)gsrc;
#endif
}

DI void wait_async_copies() {
#if USE_ASYNC_LDS
  __builtin_amdgcn_s_wait_asynccnt(0);
#endif
}

// ---------------------------------------------------------------------------
// Problem constants
// ---------------------------------------------------------------------------
#define BATCH 4
#define SEQ   2048
#define DIN   1024
#define DOUT  1024
#define BS    (BATCH * SEQ)   // 8192

// ---------------------------------------------------------------------------
// Kernel 1: f32 -> bf16 convert (weights and x)
// ---------------------------------------------------------------------------
__global__ __launch_bounds__(256) void cvt_f32_bf16(const float* __restrict__ src,
                                                    __bf16* __restrict__ dst) {
  int i = (blockIdx.x * 256 + threadIdx.x) * 4;
  float4 f = *(const float4*)(src + i);
  v4bf o;
  o[0] = (__bf16)f.x; o[1] = (__bf16)f.y; o[2] = (__bf16)f.z; o[3] = (__bf16)f.w;
  *(v4bf*)(dst + i) = o;
}

// ---------------------------------------------------------------------------
// Kernel 2: fused QKV projection: y = relu(x @ W^T + b)
//   xb : [BS][DIN] bf16 (pre-converted), staged to LDS via async copy
//   Wb : 3 x [DOUT][DIN] bf16
//   q,k: [BS][DOUT] bf16 row-major ; vT : [BATCH][DOUT][SEQ] bf16
// Workgroup tile: M=64 x N=128, 8 waves (one 16-col strip each, 4 M-tiles).
// Single K-loop: 4 A-fragments loaded once, reused across all 3 weights
// (12 WMMAs per 4 LDS-fragment loads + 3 global B loads).
// ---------------------------------------------------------------------------
#define PBM 64
#define PBN 128

__global__ __launch_bounds__(256) void qkv_proj(
    const __bf16* __restrict__ xb,
    const __bf16* __restrict__ wb,      // 3 packed weight matrices
    const float*  __restrict__ bq,
    const float*  __restrict__ bk,
    const float*  __restrict__ bv,
    __bf16* __restrict__ qout,
    __bf16* __restrict__ kout,
    __bf16* __restrict__ vtout) {
  __shared__ __bf16 xs[PBM * DIN];            // 128 KB

  const int tid   = threadIdx.x;
  const int mblk  = blockIdx.x;               // 0..127
  const int nblk  = blockIdx.y;               // 0..7
  const int srow0 = mblk * PBM;               // row in [0, BS)

  // Stage x tile into LDS with async copies (16B per lane per issue)
  {
    const __bf16* xsrc = xb + (size_t)srow0 * DIN;
#pragma unroll 4
    for (int i = tid * 8; i < PBM * DIN; i += 256 * 8)
      async_copy_b128(&xs[i], xsrc + i);
    wait_async_copies();
  }
  __syncthreads();

  const int wave = tid >> 5;
  const int lane = tid & 31;
  const int n    = lane & 15;
  const int kbA  = (lane < 16) ? 0 : 16;      // A-fragment K base
  const int kbB  = (lane < 16) ? 0 : 16;      // B-fragment K base
  const int mrow = (lane < 16) ? 0 : 8;       // C-fragment row base
  const int ocol = nblk * PBN + wave * 16 + n;  // this lane's output column

  const __bf16* wcol0 = wb + 0 * (size_t)(DOUT * DIN) + (size_t)ocol * DIN;
  const __bf16* wcol1 = wb + 1 * (size_t)(DOUT * DIN) + (size_t)ocol * DIN;
  const __bf16* wcol2 = wb + 2 * (size_t)(DOUT * DIN) + (size_t)ocol * DIN;

  v8f acc[3][4];
#pragma unroll
  for (int w = 0; w < 3; ++w)
#pragma unroll
    for (int t = 0; t < 4; ++t) { v8f z = {}; acc[w][t] = z; }

  for (int k0 = 0; k0 < DIN; k0 += 32) {
    v16bf af[4];
#pragma unroll
    for (int t = 0; t < 4; ++t)
      af[t] = load_a_frag(&xs[(t * 16 + n) * DIN + k0 + kbA]);

    v16bf bf0 = *(const v16bf*)(wcol0 + k0 + kbB);
    v16bf bf1 = *(const v16bf*)(wcol1 + k0 + kbB);
    v16bf bf2 = *(const v16bf*)(wcol2 + k0 + kbB);
#pragma unroll
    for (int t = 0; t < 4; ++t) acc[0][t] = wmma_bf16(af[t], bf0, acc[0][t]);
#pragma unroll
    for (int t = 0; t < 4; ++t) acc[1][t] = wmma_bf16(af[t], bf1, acc[1][t]);
#pragma unroll
    for (int t = 0; t < 4; ++t) acc[2][t] = wmma_bf16(af[t], bf2, acc[2][t]);
  }

  // ---- epilogue: bias + ReLU + store (q,k row-major; v transposed) ----
  {
    const float bias = bq[ocol];
#pragma unroll
    for (int t = 0; t < 4; ++t)
#pragma unroll
      for (int r = 0; r < 8; ++r) {
        float y = acc[0][t][r] + bias;
        y = y > 0.f ? y : 0.f;
        qout[(size_t)(srow0 + t * 16 + mrow + r) * DOUT + ocol] = (__bf16)y;
      }
  }
  {
    const float bias = bk[ocol];
#pragma unroll
    for (int t = 0; t < 4; ++t)
#pragma unroll
      for (int r = 0; r < 8; ++r) {
        float y = acc[1][t][r] + bias;
        y = y > 0.f ? y : 0.f;
        kout[(size_t)(srow0 + t * 16 + mrow + r) * DOUT + ocol] = (__bf16)y;
      }
  }
  {
    const float bias = bv[ocol];
    const int b  = srow0 >> 11;              // /SEQ (PBM divides SEQ)
    const int s0 = srow0 & (SEQ - 1);
    __bf16* vbase = vtout + (size_t)b * DOUT * SEQ + (size_t)ocol * SEQ;
#pragma unroll
    for (int t = 0; t < 4; ++t) {
      v8bf o;
#pragma unroll
      for (int r = 0; r < 8; ++r) {
        float y = acc[2][t][r] + bias;
        o[r] = (__bf16)(y > 0.f ? y : 0.f);
      }
      *(v8bf*)(vbase + s0 + t * 16 + mrow) = o;   // 16B contiguous store
    }
  }
}

// ---------------------------------------------------------------------------
// Kernel 3: attention for one block of 16 query rows.
//   logits(16x2048) in LDS f32, stable softmax, PV with on-the-fly
//   f32->bf16 A-fragments. 8 waves. N-tiles grouped by 4 so each
//   A-fragment is reused by 4 independent WMMAs.
// LDS: 32KB (q) + 128KB (logits) = 160KB
// ---------------------------------------------------------------------------
__global__ __launch_bounds__(256) void attn16(
    const __bf16* __restrict__ q,    // [BS][DOUT]
    const __bf16* __restrict__ k,    // [BS][DOUT]
    const __bf16* __restrict__ vt,   // [BATCH][DOUT][SEQ]
    float* __restrict__ out) {       // [BS][DOUT]
  __shared__ __bf16 qs[16 * DOUT];   // 32 KB
  __shared__ float  ls[16 * SEQ];    // 128 KB

  const int blk   = blockIdx.x;      // b*128 + qblk
  const int b     = blk >> 7;
  const int srow0 = b * SEQ + (blk & 127) * 16;
  const int tid   = threadIdx.x;
  const int wave  = tid >> 5;
  const int lane  = tid & 31;
  const int n     = lane & 15;
  const int kbA   = (lane < 16) ? 0 : 16;
  const int kbB   = (lane < 16) ? 0 : 16;
  const int mrow  = (lane < 16) ? 0 : 8;

  // stage q block (bf16, 32KB) into LDS asynchronously
  {
    const __bf16* qsrc = q + (size_t)srow0 * DOUT;
#pragma unroll
    for (int i = tid * 8; i < 16 * DOUT; i += 256 * 8)
      async_copy_b128(&qs[i], qsrc + i);
    wait_async_copies();
  }
  __syncthreads();

  // ---- logits = q @ k^T : 128 key tiles in groups of 4 per wave ----
  const __bf16* kb = k + (size_t)b * SEQ * DOUT;
  for (int g = wave; g < SEQ / 64; g += 8) {
    const int jt0 = g * 4;
    v8f acc[4];
#pragma unroll
    for (int u = 0; u < 4; ++u) { v8f z = {}; acc[u] = z; }
    const __bf16* krow[4];
#pragma unroll
    for (int u = 0; u < 4; ++u)
      krow[u] = kb + (size_t)((jt0 + u) * 16 + n) * DOUT + kbB;

    for (int k0 = 0; k0 < DOUT; k0 += 32) {
      v16bf af = load_a_frag(&qs[n * DOUT + k0 + kbA]);   // shared A frag
#pragma unroll
      for (int u = 0; u < 4; ++u) {
        v16bf bf = *(const v16bf*)(krow[u] + k0);         // contiguous 32B
        acc[u] = wmma_bf16(af, bf, acc[u]);
      }
    }
#pragma unroll
    for (int u = 0; u < 4; ++u)
#pragma unroll
      for (int r = 0; r < 8; ++r)
        ls[(mrow + r) * SEQ + (jt0 + u) * 16 + n] = acc[u][r];
  }
  __syncthreads();

  // ---- stable softmax over each of 16 rows (2 rows per wave) ----
  for (int rr = 0; rr < 2; ++rr) {
    const int row = wave * 2 + rr;
    float* L = ls + row * SEQ;
    float mx = -3.4e38f;
    for (int i = lane; i < SEQ; i += 32) mx = fmaxf(mx, L[i]);
#pragma unroll
    for (int o = 16; o >= 1; o >>= 1) mx = fmaxf(mx, __shfl_xor(mx, o, 32));
    float sum = 0.f;
    for (int i = lane; i < SEQ; i += 32) {
      float e = __expf(L[i] - mx);
      L[i] = e;
      sum += e;
    }
#pragma unroll
    for (int o = 16; o >= 1; o >>= 1) sum += __shfl_xor(sum, o, 32);
    const float inv = 1.f / sum;
    for (int i = lane; i < SEQ; i += 32) L[i] *= inv;
  }
  __syncthreads();

  // ---- out = P @ V : 64 output tiles in groups of 4 per wave ----
  const __bf16* vb = vt + (size_t)b * DOUT * SEQ;
  for (int g = wave; g < DOUT / 64; g += 8) {
    const int nt0 = g * 4;
    v8f acc[4];
#pragma unroll
    for (int u = 0; u < 4; ++u) { v8f z = {}; acc[u] = z; }
    const __bf16* vrow[4];
#pragma unroll
    for (int u = 0; u < 4; ++u)
      vrow[u] = vb + (size_t)((nt0 + u) * 16 + n) * SEQ + kbB;

    for (int j0 = 0; j0 < SEQ; j0 += 32) {
      v16bf af = load_a_frag_f32(&ls[n * SEQ + j0 + kbA]); // shared A frag
#pragma unroll
      for (int u = 0; u < 4; ++u) {
        v16bf bf = *(const v16bf*)(vrow[u] + j0);          // contiguous 32B
        acc[u] = wmma_bf16(af, bf, acc[u]);
      }
    }
#pragma unroll
    for (int u = 0; u < 4; ++u)
#pragma unroll
      for (int r = 0; r < 8; ++r)
        out[(size_t)(srow0 + mrow + r) * DOUT + (nt0 + u) * 16 + n] = acc[u][r];
  }
}

// ---------------------------------------------------------------------------
// Host launcher
// ---------------------------------------------------------------------------
extern "C" void kernel_launch(void* const* d_in, const int* in_sizes, int n_in,
                              void* d_out, int out_size, void* d_ws, size_t ws_size,
                              hipStream_t stream) {
  const float* x  = (const float*)d_in[0];
  const float* Wq = (const float*)d_in[1];
  const float* bq = (const float*)d_in[2];
  const float* Wk = (const float*)d_in[3];
  const float* bk = (const float*)d_in[4];
  const float* Wv = (const float*)d_in[5];
  const float* bv = (const float*)d_in[6];
  float* out = (float*)d_out;

  // workspace layout (bf16 elements)
  __bf16* ws   = (__bf16*)d_ws;
  const size_t WSZ = (size_t)DOUT * DIN;          // 1M per weight
  __bf16* wb   = ws;                              // 3 * WSZ
  __bf16* xbuf = ws + 3 * WSZ;                    // [BS][DIN]
  __bf16* qb   = xbuf + (size_t)BS * DIN;         // [BS][DOUT]
  __bf16* kb   = qb + (size_t)BS * DOUT;
  __bf16* vtb  = kb + (size_t)BS * DOUT;          // [BATCH][DOUT][SEQ]

  // 1) convert weights + x to bf16 (4 elems/thread)
  const int cvt_blocks_w = (DOUT * DIN) / (256 * 4);
  cvt_f32_bf16<<<cvt_blocks_w, 256, 0, stream>>>(Wq, wb + 0 * WSZ);
  cvt_f32_bf16<<<cvt_blocks_w, 256, 0, stream>>>(Wk, wb + 1 * WSZ);
  cvt_f32_bf16<<<cvt_blocks_w, 256, 0, stream>>>(Wv, wb + 2 * WSZ);
  const int cvt_blocks_x = (BS * DIN) / (256 * 4);
  cvt_f32_bf16<<<cvt_blocks_x, 256, 0, stream>>>(x, xbuf);

  // 2) fused QKV projection (+bias+ReLU), v written transposed
  qkv_proj<<<dim3(BS / PBM, DOUT / PBN), 256, 0, stream>>>(
      xbuf, wb, bq, bk, bv, qb, kb, vtb);

  // 3) attention: one block per 16 query rows
  attn16<<<BATCH * (SEQ / 16), 256, 0, stream>>>(qb, kb, vtb, out);
}
